// AxialPairAttention_76682346102981
// MI455X (gfx1250) — matrix operand order repaired and checked by reading, not compiled, not run on hardware
//
#include <hip/hip_runtime.h>
#include <hip/hip_bf16.h>

typedef __attribute__((ext_vector_type(16))) _Float16 v16h;
typedef __attribute__((ext_vector_type(8)))  _Float16 v8h;
typedef __attribute__((ext_vector_type(8)))  float    v8f;

#define NN   160
#define CC   256
#define HH   8
#define DD   32
#define BB   2
#define TOK  (BB * NN * NN)   // 51200 tokens per pass

// ---------------- WMMA fragment helpers (gfx1250 wave32 layouts) ----------------

// A-matrix 16x32 f16 fragment. base points at (row0, k0=0) of a row-major
// matrix with leading dim `ld` (elements). Lane r=l&15 is row, half=l>>4.
// elements 0..7  <- K = half*8 + 0..7
// elements 8..15 <- K = 16 + half*8 + 0..7
__device__ __forceinline__ v16h load_a_frag(const _Float16* base, int ld, int k0) {
    int lane = threadIdx.x & 31;
    const _Float16* p = base + (long)(lane & 15) * ld + k0 + ((lane >> 4) << 3);
    union { v16h v; v8h h[2]; } u;
    u.h[0] = *(const v8h*)(p);
    u.h[1] = *(const v8h*)(p + 16);
    return u.v;
}

// B-matrix 32x16 f16 fragment loaded from a "BT" source stored [n][k]
// (contiguous k, leading dim ld). Lane n=l&15 is column, half=l>>4.
// elements 0..7 <- K = half*8 + 0..7, elements 8..15 <- K = 16 + half*8 + 0..7
__device__ __forceinline__ v16h load_bt_frag(const _Float16* base, int ld, int n0, int k0) {
    int lane = threadIdx.x & 31;
    const _Float16* p = base + (long)(n0 + (lane & 15)) * ld + k0 + ((lane >> 4) << 3);
    union { v16h v; v8h h[2]; } u;
    u.h[0] = *(const v8h*)(p);
    u.h[1] = *(const v8h*)(p + 16);
    return u.v;
}

__device__ __forceinline__ v8f wmma_f16f32(v16h a, v16h b, v8f c) {
    return __builtin_amdgcn_wmma_f32_16x16x32_f16(false, a, false, b, (short)0, c, false, false);
}

// ---------------- elementwise converts ----------------

__global__ void f32_to_f16_kernel(const float* __restrict__ in, _Float16* __restrict__ out, long n) {
    long i = (long)blockIdx.x * blockDim.x + threadIdx.x;
    if (i < n) out[i] = (_Float16)in[i];
}

// w[K][N] (f32, row-major) -> wt[N][K] (f16)
__global__ void w_transpose_kernel(const float* __restrict__ w, _Float16* __restrict__ wt, int K, int N) {
    long i = (long)blockIdx.x * blockDim.x + threadIdx.x;
    if (i < (long)K * N) {
        int k = (int)(i / N);
        int n = (int)(i % N);
        wt[(long)n * K + k] = (_Float16)w[i];
    }
}

// ---------------- register-blocked WMMA GEMM ----------------
// C[M,N] = A[M,K] * B[K,N], BT stored [N][K] (contiguous K).
// Each wave computes one 16-row block x NT 16-col tiles, sharing the A fragment
// across the NT column tiles (4 independent WMMA chains hide XDL latency).

template <bool OUT_F16, int NT>
__global__ __launch_bounds__(256) void gemm_wmma_kernel(const _Float16* __restrict__ A,
                                                        const _Float16* __restrict__ BT,
                                                        void* __restrict__ Cout,
                                                        int M, int N, int K) {
    int wv   = threadIdx.x >> 5;
    int lane = threadIdx.x & 31;
    int r16  = lane & 15;
    int half = lane >> 4;

    long grp   = (long)blockIdx.x * 8 + wv;      // tile-group id
    int  gn    = N / (16 * NT);                  // groups along N
    long gtot  = (long)(M >> 4) * gn;
    if (grp >= gtot) return;
    long tm  = grp / gn;
    long tc0 = (grp % gn) * NT;

    const _Float16* a0 = A + tm * 16 * (long)K;
    const _Float16* b0[NT];
#pragma unroll
    for (int t = 0; t < NT; ++t) b0[t] = BT + (tc0 + t) * 16 * (long)K;

    v8f acc[NT];
#pragma unroll
    for (int t = 0; t < NT; ++t) acc[t] = (v8f){};

    for (int k0 = 0; k0 < K; k0 += 32) {
        if (k0 + 32 < K) {
            // pull the next A K-slab toward the WGP cache (global_prefetch_b8)
            __builtin_prefetch(a0 + (long)r16 * K + k0 + 32, 0, 0);
        }
        v16h af = load_a_frag(a0, K, k0);
#pragma unroll
        for (int t = 0; t < NT; ++t) {
            v16h bf = load_bt_frag(b0[t], K, 0, k0);
            acc[t] = wmma_f16f32(af, bf, acc[t]);
        }
    }

#pragma unroll
    for (int t = 0; t < NT; ++t) {
#pragma unroll
        for (int r = 0; r < 8; ++r) {
            long row = tm * 16 + r + 8 * half;
            long col = (tc0 + t) * 16 + r16;
            if (OUT_F16) ((_Float16*)Cout)[row * N + col] = (_Float16)acc[t][r];
            else         ((float*)Cout)[row * N + col]    = acc[t][r];
        }
    }
}

// ---------------- fused attention: per (b,m,h): S=QK^T*s+bias, softmax, O=PV ----------------
// qkv: [TOK][768] f16 (q|k|v each 256 wide, head-major inside).
// One 10-wave block per (bm, h); wave w owns S rows 16w..16w+15.

__global__ __launch_bounds__(320) void attn_kernel(const _Float16* __restrict__ qkv,
                                                   const float* __restrict__ map,   // [B][N][N]
                                                   const float* __restrict__ bw,    // [H]
                                                   const float* __restrict__ bb,    // [H]
                                                   _Float16* __restrict__ aout,     // [TOK][256]
                                                   int bias_transpose) {
    __shared__ _Float16 Pbuf[10][16 * NN];   // 51200 B : per-wave P block (f16, row-major)
    __shared__ _Float16 vT[DD * NN];         // 10240 B : V transposed [d][j]

    int bm  = blockIdx.x;          // 0..B*N-1 : (b, m)
    int h   = blockIdx.y;
    int b   = bm / NN;
    int tid = threadIdx.x;
    int wv  = tid >> 5;
    int lane = tid & 31;
    int r16  = lane & 15;
    int half = lane >> 4;

    long rowbase = (long)bm * NN;  // first token row of this (b,m) slice
    const _Float16* qbase = qkv + rowbase * 768 + h * DD;
    const _Float16* kbase = qkv + rowbase * 768 + CC + h * DD;
    const _Float16* vbase = qkv + rowbase * 768 + 2 * CC + h * DD;

    // Stage V^T into LDS: vT[d][j] = v[j][d]
    for (int j = tid; j < NN; j += blockDim.x) {
        const _Float16* vr = vbase + (long)j * 768;
#pragma unroll
        for (int d = 0; d < DD; ++d) vT[d * NN + j] = vr[d];
    }
    __syncthreads();

    int i0 = wv * 16;

    // Q fragment: rows i0..i0+15, full D=32 reduction in one WMMA K-step.
    v16h qa = load_a_frag(qbase + (long)i0 * 768, 768, 0);

    const float scale = 0.17677669529663687f;  // 1/sqrt(32)
    const float bwh = bw[h];
    const float bbh = bb[h];

    v8f acc[10];
#pragma unroll
    for (int jb = 0; jb < 10; ++jb) {
        v16h kf = load_bt_frag(kbase, 768, jb * 16, 0);  // B[d][j] = k[j][d]
        v8f z = {};
        acc[jb] = wmma_f16f32(qa, kf, z);
    }

    // scale + bias + softmax over j (160 values per row). C-layout: element r of
    // each acc lives at row i0 + r + 8*half, col jb*16 + r16. Rows for lanes 0-15
    // and 16-31 differ, so xor masks 1,2,4,8 reduce exactly within a row.
    const long mbase = (long)b * NN * NN;
#pragma unroll
    for (int r = 0; r < 8; ++r) {
        int i = i0 + r + 8 * half;
        float mmax = -3.0e38f;
#pragma unroll
        for (int jb = 0; jb < 10; ++jb) {
            int j = jb * 16 + r16;
            long midx = bias_transpose ? (mbase + (long)j * NN + i)
                                       : (mbase + (long)i * NN + j);
            float s = acc[jb][r] * scale + map[midx] * bwh + bbh;
            acc[jb][r] = s;
            mmax = fmaxf(mmax, s);
        }
#pragma unroll
        for (int m = 1; m <= 8; m <<= 1) mmax = fmaxf(mmax, __shfl_xor(mmax, m, 32));
        float ssum = 0.0f;
#pragma unroll
        for (int jb = 0; jb < 10; ++jb) {
            float e = __expf(acc[jb][r] - mmax);
            acc[jb][r] = e;
            ssum += e;
        }
#pragma unroll
        for (int m = 1; m <= 8; m <<= 1) ssum += __shfl_xor(ssum, m, 32);
        float inv = 1.0f / ssum;
        int prow = r + 8 * half;
#pragma unroll
        for (int jb = 0; jb < 10; ++jb)
            Pbuf[wv][prow * NN + jb * 16 + r16] = (_Float16)(acc[jb][r] * inv);
    }
    // Pbuf region is wave-private: no block barrier needed (DScnt handled by compiler).

    // O = P (16x160) * V (160x32): 2 column tiles, K-loop of 5 steps of 32.
    const _Float16* Pw = Pbuf[wv];
#pragma unroll
    for (int nt = 0; nt < 2; ++nt) {
        v8f oacc = {};
#pragma unroll
        for (int kb = 0; kb < 5; ++kb) {
            v16h pa = load_a_frag(Pw, NN, kb * 32);
            v16h vf = load_bt_frag(vT, NN, nt * 16, kb * 32);  // B[j][d] from vT[d][j]
            oacc = wmma_f16f32(pa, vf, oacc);
        }
#pragma unroll
        for (int r = 0; r < 8; ++r) {
            int i = i0 + r + 8 * half;
            aout[(rowbase + i) * CC + h * DD + nt * 16 + r16] = (_Float16)oacc[r];
        }
    }
}

// ---------------- residual + LayerNorm, wave per token, optional token transpose ----------------

__global__ __launch_bounds__(256) void ln_kernel(const float* __restrict__ x,
                                                 const float* __restrict__ y,
                                                 const float* __restrict__ g,
                                                 const float* __restrict__ beta,
                                                 float* __restrict__ out,
                                                 int transpose) {
    long t = (long)blockIdx.x * 8 + (threadIdx.x >> 5);
    if (t >= TOK) return;
    int lane = threadIdx.x & 31;

    const float* xr = x + t * CC;
    const float* yr = y + t * CC;
    float v[8];
    float s = 0.0f;
#pragma unroll
    for (int j = 0; j < 8; ++j) {
        v[j] = xr[lane + 32 * j] + yr[lane + 32 * j];
        s += v[j];
    }
#pragma unroll
    for (int m = 1; m < 32; m <<= 1) s += __shfl_xor(s, m, 32);
    float mean = s * (1.0f / CC);
    float vs = 0.0f;
#pragma unroll
    for (int j = 0; j < 8; ++j) {
        float d = v[j] - mean;
        vs += d * d;
    }
#pragma unroll
    for (int m = 1; m < 32; m <<= 1) vs += __shfl_xor(vs, m, 32);
    float inv = rsqrtf(vs * (1.0f / CC) + 1e-5f);

    long bb_ = t / (NN * NN);
    long rc  = t % (NN * NN);
    long rr  = rc / NN, ccol = rc % NN;
    long dt  = transpose ? (bb_ * NN * NN + ccol * NN + rr) : t;
    float* orow = out + dt * CC;
#pragma unroll
    for (int j = 0; j < 8; ++j) {
        int c = lane + 32 * j;
        orow[c] = (v[j] - mean) * inv * g[c] + beta[c];
    }
}

// ---------------- host orchestration ----------------

struct Ws {
    _Float16* xh;     // [TOK][256]
    _Float16* qkvh;   // [TOK][768]
    _Float16* aouth;  // [TOK][256]
    float*    y;      // [TOK][256]
    float*    x2;     // [TOK][256]  (pass-1 result, axis-swapped)
    _Float16* wqkvT;  // [768][256]
    _Float16* woutT;  // [256][256]
};

static void run_pass(const float* x, const float* map,
                     const float* wqkv, const float* wout,
                     const float* ln_g, const float* ln_b,
                     const float* bw, const float* bb,
                     int bias_transpose, float* dst, int dst_transpose,
                     const Ws& ws, hipStream_t stream) {
    const long NTOKC = (long)TOK * CC;

    // 1) activation convert
    f32_to_f16_kernel<<<(int)((NTOKC + 255) / 256), 256, 0, stream>>>(x, ws.xh, NTOKC);
    // 2) weight converts + transposes
    w_transpose_kernel<<<(CC * 3 * CC + 255) / 256, 256, 0, stream>>>(wqkv, ws.wqkvT, CC, 3 * CC);
    w_transpose_kernel<<<(CC * CC + 255) / 256, 256, 0, stream>>>(wout, ws.woutT, CC, CC);
    // 3) QKV GEMM: [TOK,768] = [TOK,256] x [256,768]; 4 col-tiles per wave
    {
        long groups = (long)(TOK / 16) * (3 * CC / 64);  // 3200 * 12 = 38400
        gemm_wmma_kernel<true, 4><<<(int)(groups / 8), 256, 0, stream>>>(
            ws.xh, ws.wqkvT, (void*)ws.qkvh, TOK, 3 * CC, CC);
    }
    // 4) fused attention
    attn_kernel<<<dim3(BB * NN, HH), 320, 0, stream>>>(ws.qkvh, map, bw, bb, ws.aouth, bias_transpose);
    // 5) output GEMM: [TOK,256] = [TOK,256] x [256,256]; 4 col-tiles per wave
    {
        long groups = (long)(TOK / 16) * (CC / 64);      // 3200 * 4 = 12800
        gemm_wmma_kernel<false, 4><<<(int)(groups / 8), 256, 0, stream>>>(
            ws.aouth, ws.woutT, (void*)ws.y, TOK, CC, CC);
    }
    // 6) residual + LN (+ optional token transpose on store)
    ln_kernel<<<TOK / 8, 256, 0, stream>>>(x, ws.y, ln_g, ln_b, dst, dst_transpose);
}

extern "C" void kernel_launch(void* const* d_in, const int* in_sizes, int n_in,
                              void* d_out, int out_size, void* d_ws, size_t ws_size,
                              hipStream_t stream) {
    const float* pair      = (const float*)d_in[0];
    const float* bulk_map  = (const float*)d_in[1];
    const float* row_qkv_w = (const float*)d_in[2];
    const float* row_out_w = (const float*)d_in[3];
    const float* row_ln_g  = (const float*)d_in[4];
    const float* row_ln_b  = (const float*)d_in[5];
    const float* row_bw    = (const float*)d_in[6];
    const float* row_bb    = (const float*)d_in[7];
    const float* col_qkv_w = (const float*)d_in[8];
    const float* col_out_w = (const float*)d_in[9];
    const float* col_ln_g  = (const float*)d_in[10];
    const float* col_ln_b  = (const float*)d_in[11];
    const float* col_bw    = (const float*)d_in[12];
    const float* col_bb    = (const float*)d_in[13];
    float* out = (float*)d_out;

    char* base = (char*)d_ws;
    size_t off = 0;
    auto take = [&](size_t bytes) {
        char* p = base + off;
        off = (off + bytes + 255) & ~(size_t)255;
        return p;
    };
    Ws ws;
    ws.xh    = (_Float16*)take((size_t)TOK * CC * 2);
    ws.qkvh  = (_Float16*)take((size_t)TOK * 3 * CC * 2);
    ws.aouth = (_Float16*)take((size_t)TOK * CC * 2);
    ws.y     = (float*)take((size_t)TOK * CC * 4);
    ws.x2    = (float*)take((size_t)TOK * CC * 4);
    ws.wqkvT = (_Float16*)take((size_t)3 * CC * CC * 2);
    ws.woutT = (_Float16*)take((size_t)CC * CC * 2);

    // Pass 1 (row attention): result written axis-swapped into x2, so x2 == xt.
    run_pass(pair, bulk_map, row_qkv_w, row_out_w, row_ln_g, row_ln_b,
             row_bw, row_bb, /*bias_transpose=*/0, ws.x2, /*dst_transpose=*/1,
             ws, stream);
    // Pass 2 (column attention on xt): bias uses map^T; final LN writes the
    // swap-back directly into d_out.
    run_pass(ws.x2, bulk_map, col_qkv_w, col_out_w, col_ln_g, col_ln_b,
             col_bw, col_bb, /*bias_transpose=*/1, out, /*dst_transpose=*/1,
             ws, stream);
}
